// PointNetFeaturePropagation_68186900791887
// MI455X (gfx1250) — matrix operand order repaired and checked by reading, not compiled, not run on hardware
//
#include <hip/hip_runtime.h>
#include <hip/hip_bf16.h>

typedef __attribute__((ext_vector_type(16))) _Float16 v16h;
typedef __attribute__((ext_vector_type(8)))  float    v8f;

#define T_  4
#define B_  2
#define N_  4096
#define S_  1024
#define D_  128
#define M0_ 256
#define M1_ 128
#define MTOT  (T_*B_*N_)            /* 32768 rows of the GEMMs      */
#define BND   ((size_t)B_*N_*D_)    /* 1,048,576 per-timestep elems */
#define BNO   ((size_t)B_*N_*M0_)   /* 2,097,152 per-timestep elems */

// ---------------------------------------------------------------------------
// f32 -> f16 weight conversion
// ---------------------------------------------------------------------------
__global__ __launch_bounds__(256) void w2h_kernel(const float* __restrict__ w,
                                                  _Float16* __restrict__ wh, int n) {
    int i = blockIdx.x * 256 + threadIdx.x;
    if (i < n) wh[i] = (_Float16)w[i];
}

// ---------------------------------------------------------------------------
// 3-NN inverse-distance interpolation:  X0 = points1 + sum_i w_i * points2[idx_i]
// grid: (N/256, T*B), block 256.  xyz2 tile staged in LDS (12 KB).
// ---------------------------------------------------------------------------
__global__ __launch_bounds__(256) void interp_kernel(const float* __restrict__ xyz1,
                                                     const float* __restrict__ xyz2,
                                                     const float* __restrict__ p1,
                                                     const float* __restrict__ p2,
                                                     float* __restrict__ X0) {
    __shared__ float s2[S_ * 3];
    const int tb = blockIdx.y;
    const int n  = blockIdx.x * 256 + threadIdx.x;

    const float* x2 = xyz2 + (size_t)tb * S_ * 3;
    for (int i = threadIdx.x; i < S_ * 3; i += 256) s2[i] = x2[i];
    __syncthreads();

    const float* x1 = xyz1 + ((size_t)tb * N_ + n) * 3;
    const float ax = x1[0], ay = x1[1], az = x1[2];

    float d0 = 3.4e38f, d1 = 3.4e38f, d2 = 3.4e38f;
    int   i0 = 0,       i1 = 0,       i2 = 0;
    for (int s = 0; s < S_; ++s) {
        float dx = ax - s2[3*s+0];
        float dy = ay - s2[3*s+1];
        float dz = az - s2[3*s+2];
        float d  = dx*dx + dy*dy + dz*dz;
        if (d < d0)      { d2=d1; i2=i1; d1=d0; i1=i0; d0=d; i0=s; }
        else if (d < d1) { d2=d1; i2=i1; d1=d;  i1=s; }
        else if (d < d2) { d2=d;  i2=s; }
    }
    const float r0 = 1.0f / (d0 + 1e-8f);
    const float r1 = 1.0f / (d1 + 1e-8f);
    const float r2 = 1.0f / (d2 + 1e-8f);
    const float rs = 1.0f / (r0 + r1 + r2);
    const float w0 = r0 * rs, w1 = r1 * rs, w2 = r2 * rs;

    const float4* P2  = (const float4*)(p2 + (size_t)tb * S_ * D_);
    const float4* P1  = (const float4*)(p1 + ((size_t)tb * N_ + n) * D_);
    float4*       out = (float4*)(X0 + ((size_t)tb * N_ + n) * D_);
    const int q0 = i0 * (D_/4), q1 = i1 * (D_/4), q2 = i2 * (D_/4);
    #pragma unroll 4
    for (int q = 0; q < D_/4; ++q) {
        float4 a = P2[q0 + q], b = P2[q1 + q], c = P2[q2 + q], p = P1[q];
        float4 r;
        r.x = p.x + w0*a.x + w1*b.x + w2*c.x;
        r.y = p.y + w0*a.y + w1*b.y + w2*c.y;
        r.z = p.z + w0*a.z + w1*b.z + w2*c.z;
        r.w = p.w + w0*a.w + w1*b.w + w2*c.w;
        out[q] = r;
    }
}

// ---------------------------------------------------------------------------
// LIF scan over T on f32 input, emit binary spikes packed as f16 (exact in f16)
// ---------------------------------------------------------------------------
__global__ __launch_bounds__(256) void lif_pack_kernel(const float* __restrict__ X,
                                                       _Float16* __restrict__ Sp,
                                                       size_t stride) {
    const size_t j = (size_t)blockIdx.x * 256 + threadIdx.x;
    float v = 0.0f;
    #pragma unroll
    for (int t = 0; t < T_; ++t) {
        float x = X[(size_t)t * stride + j];
        float h = v + (x - v) * 0.5f;          // tau = 2
        float s = (h >= 0.5f) ? 1.0f : 0.0f;   // v_th = 0.5, hard reset
        v = h * (1.0f - s);
        Sp[(size_t)t * stride + j] = (_Float16)s;
    }
}

// ---------------------------------------------------------------------------
// Fused BN-affine + LIF + f16 pack (between layer 0 and layer 1)
// aff[o] = scale, aff[O+o] = shift
// ---------------------------------------------------------------------------
__global__ __launch_bounds__(256) void bn_lif_pack_kernel(const float* __restrict__ Y,
                                                          const float* __restrict__ aff,
                                                          _Float16* __restrict__ Sp,
                                                          int O, size_t stride) {
    const size_t j = (size_t)blockIdx.x * 256 + threadIdx.x;
    const int o = (int)(j & (size_t)(O - 1));
    const float sc = aff[o], sh = aff[O + o];
    float v = 0.0f;
    #pragma unroll
    for (int t = 0; t < T_; ++t) {
        float x = Y[(size_t)t * stride + j] * sc + sh;
        float h = v + (x - v) * 0.5f;
        float s = (h >= 0.5f) ? 1.0f : 0.0f;
        v = h * (1.0f - s);
        Sp[(size_t)t * stride + j] = (_Float16)s;
    }
}

// ---------------------------------------------------------------------------
// WMMA GEMM:  Y[m,o] = sum_k Sp[m,k] * Wh[o,k] + bias[o]
// block = 256 threads = 8 waves; wave w computes 16x16 tile at
// (blockIdx.x*128 + w*16, blockIdx.y*16), K-loop in steps of 32.
// The 16xK weight tile (shared by all 8 waves) is DMA'd into LDS once with
// gfx1250 async-to-LDS loads (ASYNCcnt), rows padded by 8 halves to spread
// the 16 B-fragment lanes across LDS banks.  Fragment layouts follow
// CDNA5 ISA §7.12.2.
// ---------------------------------------------------------------------------
template <int K, int O>
__global__ __launch_bounds__(256) void gemm_wmma_kernel(const _Float16* __restrict__ Sp,
                                                        const _Float16* __restrict__ Wh,
                                                        const float* __restrict__ bias,
                                                        float* __restrict__ Y) {
    constexpr int KP = K + 8;                  // padded row stride (halves)
    __shared__ _Float16 Bs[16 * KP];           // weight tile [16 o][K k]

    const int lane   = threadIdx.x & 31;
    const int wave   = threadIdx.x >> 5;
    const int laneLo = lane & 15;
    const int laneHi = lane >> 4;
    const int m0   = blockIdx.x * 128 + wave * 16;
    const int o0   = blockIdx.y * 16;
    const int mrow = m0 + laneLo;              // A row owned by this lane
    const int ocol = o0 + laneLo;              // B column owned by this lane

    // --- async-DMA the weight tile (contiguous 16*K halves in Wh) into LDS ---
    constexpr int CHUNKS = 16 * K / 8;         // 16-byte chunks
    const _Float16* gbase = Wh + (size_t)o0 * K;
    #pragma unroll
    for (int it = 0; it < (CHUNKS + 255) / 256; ++it) {
        const int c   = it * 256 + threadIdx.x;
        const int row = c / (K / 8);
        const int col = (c % (K / 8)) * 8;
        const _Float16* g = gbase + c * 8;
        const unsigned  l = (unsigned)(size_t)(const void*)&Bs[row * KP + col];
        asm volatile("global_load_async_to_lds_b128 %0, %1, off"
                     :: "v"(l), "v"(g) : "memory");
    }
    asm volatile("s_wait_asynccnt 0x0" ::: "memory");
    __syncthreads();

    const unsigned* Arow  = (const unsigned*)(Sp + (size_t)mrow * K);
    const unsigned* BrowL = (const unsigned*)(&Bs[laneLo * KP]);

    v8f acc = {};
    #pragma unroll
    for (int k0 = 0; k0 < K; k0 += 32) {
        union { v16h h; unsigned u[8]; } A, Bf;
        #pragma unroll
        for (int v = 0; v < 8; ++v) {
            // A (16-bit 16x32): lanes 0-15 hold K 0-7 & 16-23, lanes 16-31 hold 8-15 & 24-31
            const int ka = k0 + ((v & 3) * 2) + laneHi * 8 + ((v >> 2) * 16);
            A.u[v]  = Arow[ka >> 1];
            // B (16-bit 32x16): n = lane%16, same K order as A
            const int kb = k0 + ((v & 3) * 2) + laneHi * 8 + ((v >> 2) * 16);
            Bf.u[v] = BrowL[kb >> 1];
        }
        acc = __builtin_amdgcn_wmma_f32_16x16x32_f16(
            /*neg_a=*/false, A.h, /*neg_b=*/false, Bf.h,
            /*c_mod=*/(short)0, acc, /*reuse_a=*/false, /*reuse_b=*/false);
    }

    const float bo = bias[ocol];
    #pragma unroll
    for (int v = 0; v < 8; ++v) {
        // D (f32 16x16): M = v + 8*(lane/16), N = lane%16
        const int m = m0 + v + laneHi * 8;
        Y[(size_t)m * O + ocol] = acc[v] + bo;
    }
}

// ---------------------------------------------------------------------------
// BatchNorm batch statistics -> affine (scale, shift) per channel.
// One block per output channel o; reduce over MTOT rows.
// ---------------------------------------------------------------------------
__global__ __launch_bounds__(256) void bn_stats_kernel(const float* __restrict__ Y,
                                                       const float* __restrict__ g,
                                                       const float* __restrict__ b,
                                                       float* __restrict__ aff, int O) {
    __shared__ float ss[256], sq[256];
    const int o = blockIdx.x;
    float s = 0.0f, q = 0.0f;
    for (int m = threadIdx.x; m < MTOT; m += 256) {
        float y = Y[(size_t)m * O + o];
        s += y; q += y * y;
    }
    ss[threadIdx.x] = s; sq[threadIdx.x] = q;
    __syncthreads();
    for (int off = 128; off > 0; off >>= 1) {
        if (threadIdx.x < off) {
            ss[threadIdx.x] += ss[threadIdx.x + off];
            sq[threadIdx.x] += sq[threadIdx.x + off];
        }
        __syncthreads();
    }
    if (threadIdx.x == 0) {
        const float inv = 1.0f / (float)MTOT;
        const float mean = ss[0] * inv;
        const float var  = sq[0] * inv - mean * mean;
        const float sc   = g[o] * rsqrtf(var + 1e-5f);
        aff[o]     = sc;
        aff[O + o] = b[o] - mean * sc;
    }
}

// ---------------------------------------------------------------------------
// Final BN apply -> d_out
// ---------------------------------------------------------------------------
__global__ __launch_bounds__(256) void bn_apply_kernel(const float* __restrict__ Y,
                                                       const float* __restrict__ aff,
                                                       float* __restrict__ out, int O) {
    const size_t i = (size_t)blockIdx.x * 256 + threadIdx.x;
    const int o = (int)(i & (size_t)(O - 1));
    out[i] = Y[i] * aff[o] + aff[O + o];
}

// ---------------------------------------------------------------------------
extern "C" void kernel_launch(void* const* d_in, const int* in_sizes, int n_in,
                              void* d_out, int out_size, void* d_ws, size_t ws_size,
                              hipStream_t stream) {
    const float* xyz1    = (const float*)d_in[0];
    const float* xyz2    = (const float*)d_in[1];
    const float* points1 = (const float*)d_in[2];
    const float* points2 = (const float*)d_in[3];
    const float* conv_w0 = (const float*)d_in[4];
    const float* conv_b0 = (const float*)d_in[5];
    const float* bn_g0   = (const float*)d_in[6];
    const float* bn_b0   = (const float*)d_in[7];
    const float* conv_w1 = (const float*)d_in[8];
    const float* conv_b1 = (const float*)d_in[9];
    const float* bn_g1   = (const float*)d_in[10];
    const float* bn_b1   = (const float*)d_in[11];

    char* ws = (char*)d_ws;
    float*    X0   = (float*)(ws);                             // 16 MB
    _Float16* Sp   = (_Float16*)(ws + (16u << 20));            // 16 MB (max K=256)
    float*    Y    = (float*)(ws + (32u << 20));               // 32 MB (max O=256)
    _Float16* Wh0  = (_Float16*)(ws + (64u << 20));            // 64 KB
    _Float16* Wh1  = (_Float16*)(ws + (64u << 20) + 65536);    // 64 KB
    float*    aff0 = (float*)(ws + (64u << 20) + 2 * 65536);   // 2 KB
    float*    aff1 = (float*)(ws + (64u << 20) + 2 * 65536 + 2048);

    // weights -> f16
    w2h_kernel<<<(M0_*D_ + 255) / 256, 256, 0, stream>>>(conv_w0, Wh0, M0_ * D_);
    w2h_kernel<<<(M1_*M0_ + 255) / 256, 256, 0, stream>>>(conv_w1, Wh1, M1_ * M0_);

    // 3-NN interpolation -> X0 [T,B,N,128]
    interp_kernel<<<dim3(N_ / 256, T_ * B_), 256, 0, stream>>>(xyz1, xyz2, points1, points2, X0);

    // ---- layer 0 ----
    lif_pack_kernel<<<(unsigned)(BND / 256), 256, 0, stream>>>(X0, Sp, BND);
    gemm_wmma_kernel<D_, M0_><<<dim3(MTOT / 128, M0_ / 16), 256, 0, stream>>>(Sp, Wh0, conv_b0, Y);
    bn_stats_kernel<<<M0_, 256, 0, stream>>>(Y, bn_g0, bn_b0, aff0, M0_);

    // ---- layer 1 ----
    bn_lif_pack_kernel<<<(unsigned)(BNO / 256), 256, 0, stream>>>(Y, aff0, Sp, M0_, BNO);
    gemm_wmma_kernel<M0_, M1_><<<dim3(MTOT / 128, M1_ / 16), 256, 0, stream>>>(Sp, Wh1, conv_b1, Y);
    bn_stats_kernel<<<M1_, 256, 0, stream>>>(Y, bn_g1, bn_b1, aff1, M1_);

    bn_apply_kernel<<<(unsigned)((size_t)MTOT * M1_ / 256), 256, 0, stream>>>(Y, aff1, (float*)d_out, M1_);
}
